// LSTM_network_39273180954672
// MI455X (gfx1250) — matrix operand order repaired and checked by reading, not compiled, not run on hardware
//
#include <hip/hip_runtime.h>

typedef __attribute__((ext_vector_type(16))) _Float16 v16h;
typedef __attribute__((ext_vector_type(8)))  _Float16 v8h;
typedef __attribute__((ext_vector_type(8)))  float    v8f;

#define B_    512
#define T_    2048
#define I_    13
#define H_    13
#define ROWS  16     // batch rows per wave (WMMA M)
#define KP    32     // K padded: [x(13) | h(13) | 0(6)]
#define NP    64     // gate columns padded: 52 gates + fc row (52) + pad
#define NWORK 208    // 16 rows * 13 hidden
#define PFD   8      // x prefetch distance (steps) via global_prefetch_b8

// Native v_tanh_f32 when available (CDNA5 trans op); exp2+rcp fallback otherwise.
#if __has_builtin(__builtin_amdgcn_tanhf)
  #define TANH_NATIVE __builtin_amdgcn_tanhf
#elif __has_builtin(__builtin_amdgcn_tanh_f32)
  #define TANH_NATIVE __builtin_amdgcn_tanh_f32
#endif

__device__ __forceinline__ float tanh_fast(float x) {
#ifdef TANH_NATIVE
  return TANH_NATIVE(x);
#else
  return 2.0f * __builtin_amdgcn_rcpf(1.0f + __builtin_amdgcn_exp2f(-2.8853900817779268f * x)) - 1.0f;
#endif
}
__device__ __forceinline__ float sigmoid_fast(float x) {
  return 0.5f * tanh_fast(0.5f * x) + 0.5f;   // 1 trans op instead of exp+rcp
}

__global__ __launch_bounds__(32)
void lstm_wmma_kernel(const float* __restrict__ x,
                      const float* __restrict__ Wih,
                      const float* __restrict__ Whh,
                      const float* __restrict__ bih,
                      const float* __restrict__ bhh,
                      const float* __restrict__ fcw,
                      const float* __restrict__ fcb,
                      float* __restrict__ out)
{
  // A-operand staging: act[row][k], k = [x(0..12) | h(13..25) | zero pad].
  // Row 16 is a dummy sink so padding lanes can store unconditionally (never read).
  __shared__ __align__(32) _Float16 actA[ROWS + 1][KP];
  // D tiles, stored transposed: G[gate][row]
  __shared__ __align__(32) float    G[NP][ROWS];

  const int lane  = threadIdx.x;   // 0..31
  const int ncol  = lane & 15;
  const int khalf = lane >> 4;     // which K-half this lane holds (ISA 16-bit A/B layout)
  const int rb    = blockIdx.x * ROWS;

  // ---- Build register-resident B operands (combined [W_ih | W_hh | 0] + fc row) ----
  // B layout (16-bit, 32x16): lane<16 holds K=0..7 & 16..23 of column N=lane, lane>=16 holds K=8..15 & 24..31.
  v16h bt[4];
  v8f  cbv[4];   // per-column bias broadcast, pinned live in VGPRs across the whole loop
#pragma unroll
  for (int n = 0; n < 4; ++n) {
    const int g = n * 16 + ncol;   // gate row index (0..51 gates, 52 = fc head)
    v16h bv;
#pragma unroll
    for (int kk = 0; kk < 16; ++kk) {
      const int k = (kk < 8) ? (khalf * 8 + kk) : (16 + khalf * 8 + (kk - 8));
      float w = 0.0f;
      if (g < 52) {
        if (k < 13)       w = Wih[g * 13 + k];
        else if (k < 26)  w = Whh[g * 13 + (k - 13)];
      } else if (g == 52 && k >= 13 && k < 26) {
        w = fcw[k - 13];           // fc folded in as gate row 52 (h-columns only)
      }
      bv[kk] = (_Float16)w;
    }
    bt[n] = bv;
    float b = 0.0f;
    if (g < 52)       b = bih[g] + bhh[g];
    else if (g == 52) b = fcb[0];
#pragma unroll
    for (int q = 0; q < 8; ++q) cbv[n][q] = b;
  }
  // Pin the bias vectors so the compiler can't rematerialize the broadcast per step.
#pragma unroll
  for (int n = 0; n < 4; ++n)
    asm volatile("" : "+v"(cbv[n]));

  // zero the activation panel (h-part + pad start at 0; x-part rewritten each step)
  for (int i = lane; i < (ROWS + 1) * KP; i += 32)
    (&actA[0][0])[i] = (_Float16)0.0f;

  // ---- static (row,h) work map for the elementwise phase: 208 items over 32 lanes x 7 ----
  // Padding slots (idx >= 208) map to dummy row 16 and a clamped x pointer: fully branch-free.
  int   mI[7], hI[7];
  float creg[7];              // cell state lives in registers
  const float* xp[7];         // advancing x pointers (one element per slot per step)
#pragma unroll
  for (int j = 0; j < 7; ++j) {
    int idx = j * 32 + lane;
    bool v  = idx < NWORK;
    int m   = idx / 13;
    int h   = idx - m * 13;
    if (!v) { m = ROWS; h = 0; }           // dummy sink row
    mI[j] = m; hI[j] = h; creg[j] = 0.0f;
    xp[j] = v ? (x + (size_t)(rb + m) * ((size_t)T_ * I_) + h) : x;
  }
  float* op = out + (size_t)(rb + ncol) * T_;   // lanes<16 write the output column

  // preload x_0
  float xr[7];
#pragma unroll
  for (int j = 0; j < 7; ++j) xr[j] = *xp[j];

  __syncthreads();

  // ---- main recurrence: T+1 iterations (last one only flushes fc(h_{T-1})) ----
  for (int t = 0; t <= T_; ++t) {
    if (t < T_) {
#pragma unroll
      for (int j = 0; j < 7; ++j)
        actA[mI[j]][hI[j]] = (_Float16)xr[j];
    }
    __syncthreads();

    // A operand: lane<16 holds (M=lane, K 0..7 & 16..23), lane>=16 (M=lane-16, K 8..15 & 24..31)
    const _Float16* ap = &actA[ncol][0] + khalf * 8;
    v8h a0 = *(const v8h*)(ap);
    v8h a1 = *(const v8h*)(ap + 16);
    v16h a = __builtin_shufflevector(a0, a1, 0,1,2,3,4,5,6,7,8,9,10,11,12,13,14,15);

    // register prefetch of x_{t+1} (hits cache thanks to the deep prefetch below)
    if (t + 1 < T_) {
#pragma unroll
      for (int j = 0; j < 7; ++j) {
        xp[j] += I_;
        xr[j] = *xp[j];
      }
    }
    // deep prefetch: global_prefetch_b8 for x_{t+PFD}; no LOADcnt, so barriers ignore it
    if (t + PFD < T_) {
#pragma unroll
      for (int j = 0; j < 7; ++j)
        __builtin_prefetch(xp[j] + (PFD - 1) * I_, 0, 1);
    }

    // gates[16 rows x 64 cols] = [x_t ; h] * W^T + bias  (4 WMMAs, weights + bias in VGPRs)
#pragma unroll
    for (int n = 0; n < 4; ++n) {
      v8f d = __builtin_amdgcn_wmma_f32_16x16x32_f16(
          false, a, false, bt[n], (short)0, cbv[n], false, false);
      // D layout: lane holds column n=16n+ncol, rows m = khalf*8 .. +7  -> G[col][m] contiguous
      *(v8f*)&G[n * 16 + ncol][khalf * 8] = d;
    }
    __syncthreads();

    // output head: column 52 == dot(fc_w, h_{t-1}) + fc_b
    if (t > 0 && lane < 16)
      op[t - 1] = sigmoid_fast(G[52][lane]);

    if (t < T_) {
      // elementwise LSTM update; PyTorch gate order i,f,g,o in rows 0..51
#pragma unroll
      for (int j = 0; j < 7; ++j) {
        const int m = mI[j], h = hI[j];
        float ig = G[h     ][m & 15];      // padding slots read discarded garbage
        float fg = G[13 + h][m & 15];
        float gg = G[26 + h][m & 15];
        float og = G[39 + h][m & 15];
        float cn = sigmoid_fast(fg) * creg[j] + sigmoid_fast(ig) * tanh_fast(gg);
        creg[j]  = cn;
        float hn = sigmoid_fast(og) * tanh_fast(cn);
        actA[m][13 + h] = (_Float16)hn;    // valid -> next step's A operand; padding -> row 16
      }
    }
    __syncthreads();
  }
}

extern "C" void kernel_launch(void* const* d_in, const int* in_sizes, int n_in,
                              void* d_out, int out_size, void* d_ws, size_t ws_size,
                              hipStream_t stream) {
  const float* x   = (const float*)d_in[0];
  const float* Wih = (const float*)d_in[1];
  const float* Whh = (const float*)d_in[2];
  const float* bih = (const float*)d_in[3];
  const float* bhh = (const float*)d_in[4];
  const float* fcw = (const float*)d_in[5];
  const float* fcb = (const float*)d_in[6];
  float* out = (float*)d_out;

  dim3 grid(B_ / ROWS);   // 32 workgroups, one wave each (16 batch rows per wave)
  dim3 block(32);
  lstm_wmma_kernel<<<grid, block, 0, stream>>>(x, Wih, Whh, bih, bhh, fcw, fcb, out);
}
